// BDLayer_45715631899545
// MI455X (gfx1250) — compile-verified, matrix operand order checked by reading, not run on hardware
//
#include <hip/hip_runtime.h>
#include <hip/hip_bf16.h>

#define F_IN   128
#define F_OUT  64
#define LRELU  0.01f
#define BN_EPS 1e-5f

// padded LDS row stride (halfwords): 128 + 8 -> 272 bytes = 68 dwords,
// 68 mod 64 = 4 -> conflict-free ds_load_b128 across 16 lanes
#define LSTRIDE  136
#define LSTRIDEU 68

typedef __attribute__((ext_vector_type(16))) __bf16 v16bf;
typedef __attribute__((ext_vector_type(8)))  float  v8f;

__device__ __forceinline__ unsigned int pack_bf16x2(float2 w) {
    __hip_bfloat162 r = __float22bfloat162_rn(w);   // packed RNE cvt
    return *(unsigned int*)&r;
}

// ---------------------------------------------------------------------------
// Kernel 1: h = x @ W^T via v_wmma_f32_16x16x32_bf16.
// Block = 256 threads = 8 waves covering 32 rows x 64 cols.
// W (64x128) and the 32x128 x-tile are staged in LDS as bf16 once per block;
// each wave assembles A/B fragments with 2x ds_load_b128 per matrix per K-step.
// ---------------------------------------------------------------------------
__global__ __launch_bounds__(256) void gemm_wmma_bf16(
    const float* __restrict__ x, const float* __restrict__ W,
    float* __restrict__ h, int n_nodes)
{
    __shared__ unsigned short Wlds[F_OUT * LSTRIDE];   // ~17 KB
    __shared__ unsigned short Xlds[32 * LSTRIDE];      // ~8.5 KB
    const int tid = threadIdx.x;

    // Stage W: 64x128 f32 -> bf16 (4096 f32 pairs, 16 per thread)
    {
        const float2* W2 = (const float2*)W;
        unsigned int* Wu = (unsigned int*)Wlds;
        for (int i = tid; i < (F_OUT * F_IN) / 2; i += 256) {
            int row = i >> 6, colp = i & 63;           // 64 pairs per row
            Wu[row * LSTRIDEU + colp] = pack_bf16x2(W2[i]);
        }
    }
    // Stage x tile: 32 rows x 128 f32 -> bf16 (16 consecutive f32 per thread)
    {
        const int m0  = blockIdx.x * 32;
        unsigned int* Xu = (unsigned int*)Xlds;
        const int off = tid * 16;                      // 0..4095
        const int rl  = off >> 7;                      // local row
        const int col = off & 127;
        int row = m0 + rl;
        if (row >= n_nodes) row = n_nodes - 1;         // clamp (masked at store)
        const float2* src = (const float2*)(x + (size_t)row * F_IN + col);
        unsigned int* dst = Xu + rl * LSTRIDEU + (col >> 1);
#pragma unroll
        for (int j = 0; j < 8; ++j) dst[j] = pack_bf16x2(src[j]);
    }
    __syncthreads();

    const int wave = tid >> 5;
    const int lane = tid & 31;
    const int half = lane >> 4;                        // ISA half-wave split
    const int lm   = lane & 15;

    const int m_base = blockIdx.x * 32 + (wave >> 2) * 16;
    const int n_base = (wave & 3) * 16;
    const int arow_l = (wave >> 2) * 16 + lm;          // row in Xlds
    const int bcol   = n_base + lm;                    // row of W = col of B

    union frag { v16bf v; uint4 q[2]; };
    v8f acc = {};

#pragma unroll
    for (int kt = 0; kt < 4; ++kt) {
        // per ISA 16-bit layout: lane holds K = kb..kb+7 and kb+16..kb+23
        const int kb = kt * 32 + (half << 3);
        frag A, B;
        A.q[0] = *(const uint4*)(&Xlds[arow_l * LSTRIDE + kb]);
        A.q[1] = *(const uint4*)(&Xlds[arow_l * LSTRIDE + kb + 16]);
        B.q[0] = *(const uint4*)(&Wlds[bcol   * LSTRIDE + kb]);
        B.q[1] = *(const uint4*)(&Wlds[bcol   * LSTRIDE + kb + 16]);
        acc = __builtin_amdgcn_wmma_f32_16x16x32_bf16(
            /*neg_a=*/false, A.v, /*neg_b=*/false, B.v,
            /*c_mod=*/(short)0, acc, /*reuse_a=*/false, /*reuse_b=*/false);
    }

    // D layout: lane l, reg r -> m = r + 8*(l/16), n = l%16.
    // 32-bit offsets (n*64 = 6.4M << 2^31); uniform fast path when the whole
    // tile is in range (always true for N divisible by 32 -> no divergence).
    const int base_idx = (m_base + (half << 3)) * F_OUT + n_base + lm;
    if (m_base + 16 <= n_nodes) {
#pragma unroll
        for (int r = 0; r < 8; ++r) h[base_idx + r * F_OUT] = acc[r];
    } else {
#pragma unroll
        for (int r = 0; r < 8; ++r) {
            if (m_base + r + (half << 3) < n_nodes) h[base_idx + r * F_OUT] = acc[r];
        }
    }
}

// ---------------------------------------------------------------------------
// Kernel 2a/2b: degree (self loop contributes 1), then atomic count of targets
// ---------------------------------------------------------------------------
__global__ void deg_init(float* __restrict__ deg, int n) {
    int i = blockIdx.x * blockDim.x + threadIdx.x;
    if (i < n) deg[i] = 1.0f;                          // self loop
}

__global__ void deg_count(const long long* __restrict__ tgt,
                          float* __restrict__ deg, int E) {
    int e = blockIdx.x * blockDim.x + threadIdx.x;
    if (e < E) atomicAdd(&deg[(int)tgt[e]], 1.0f);
}

// ---------------------------------------------------------------------------
// Kernel 3: dinv = rsqrt(deg); agg initialized with self-loop term h*dinv^2
// ---------------------------------------------------------------------------
__global__ void selfloop_init(const float* __restrict__ h,
                              const float* __restrict__ deg,
                              float* __restrict__ dinv,
                              float* __restrict__ agg, int n) {
    int idx = blockIdx.x * blockDim.x + threadIdx.x;   // < n*64 = 6.4M
    if (idx >= n * F_OUT) return;
    int node = idx >> 6;
    float di = rsqrtf(deg[node]);                      // deg >= 1 always
    if ((idx & 63) == 0) dinv[node] = di;
    agg[idx] = h[idx] * di * di;
}

// ---------------------------------------------------------------------------
// Kernel 4: edge scatter. 64 consecutive lanes = one edge's 64 features
// (contiguous 256B gather of h[src], 256B atomic scatter into agg[tgt];
//  agg is fully L2-resident: 25.6 MB << 192 MB).
// ---------------------------------------------------------------------------
__global__ __launch_bounds__(256) void edge_scatter(
    const long long* __restrict__ ei, const float* __restrict__ h,
    const float* __restrict__ dinv, float* __restrict__ agg, int E)
{
    long idx = (long)blockIdx.x * blockDim.x + threadIdx.x;
    long e = idx >> 6;
    if (e >= E) return;
    int f = (int)(idx & 63);
    int s = (int)ei[e];
    int t = (int)ei[(long)E + e];
    float coef = dinv[s] * dinv[t];
    atomicAdd(&agg[t * F_OUT + f], h[s * F_OUT + f] * coef);
}

// ---------------------------------------------------------------------------
// Kernel 5: mean-aggregation + bias + LeakyReLU, accumulate BN statistics.
// 128 nodes/block; register partials -> LDS reduce -> 64 atomics per block.
// act aliases agg (elementwise in-place).
// ---------------------------------------------------------------------------
__global__ __launch_bounds__(256) void finalize_stats(
    float* __restrict__ agg_act, const float* __restrict__ deg,
    const float* __restrict__ bias, float* __restrict__ fsum,
    float* __restrict__ fss, int n)
{
    const int f   = threadIdx.x & 63;
    const int sub = threadIdx.x >> 6;                  // 0..3
    const int base = blockIdx.x * 128;
    const float b = bias[f];

    float s = 0.f, ss = 0.f;
    for (int j = sub; j < 128; j += 4) {
        int node = base + j;
        if (node >= n) break;
        int idx = node * F_OUT + f;
        float v = agg_act[idx] / fmaxf(deg[node], 1.0f) + b;
        v = v > 0.f ? v : LRELU * v;                   // LeakyReLU
        agg_act[idx] = v;
        s += v; ss += v * v;
    }

    __shared__ float ls[256], lss[256];
    ls[threadIdx.x] = s; lss[threadIdx.x] = ss;
    __syncthreads();
    if (sub == 0) {
        s  = ls[f]  + ls[64 + f]  + ls[128 + f]  + ls[192 + f];
        ss = lss[f] + lss[64 + f] + lss[128 + f] + lss[192 + f];
        atomicAdd(&fsum[f], s);
        atomicAdd(&fss[f], ss);
    }
}

__global__ void zero_stats(float* __restrict__ fsum, float* __restrict__ fss) {
    if (threadIdx.x < 64) { fsum[threadIdx.x] = 0.f; fss[threadIdx.x] = 0.f; }
}

// ---------------------------------------------------------------------------
// Kernel 6: BatchNorm apply (biased batch stats, training mode)
// ---------------------------------------------------------------------------
__global__ void bn_apply(const float* __restrict__ act,
                         const float* __restrict__ fsum,
                         const float* __restrict__ fss,
                         const float* __restrict__ gamma,
                         const float* __restrict__ beta,
                         float* __restrict__ y, int n)
{
    int idx = blockIdx.x * blockDim.x + threadIdx.x;   // < n*64 = 6.4M
    if (idx >= n * F_OUT) return;
    int f = idx & 63;
    float inv_n = 1.0f / (float)n;
    float mu  = fsum[f] * inv_n;
    float var = fss[f] * inv_n - mu * mu;
    y[idx] = (act[idx] - mu) * rsqrtf(var + BN_EPS) * gamma[f] + beta[f];
}

// ---------------------------------------------------------------------------
extern "C" void kernel_launch(void* const* d_in, const int* in_sizes, int n_in,
                              void* d_out, int out_size, void* d_ws, size_t ws_size,
                              hipStream_t stream)
{
    const float*     x     = (const float*)d_in[0];
    const long long* ei    = (const long long*)d_in[1];   // int64 [2, E] flat
    const float*     W     = (const float*)d_in[2];
    const float*     bias  = (const float*)d_in[3];
    const float*     gamma = (const float*)d_in[4];
    const float*     beta  = (const float*)d_in[5];
    float*           y     = (float*)d_out;

    const int n = in_sizes[0] / F_IN;      // 100000
    const int E = in_sizes[1] / 2;         // 1600000

    // workspace layout (floats), ~52 MB total
    float* h    = (float*)d_ws;                 // n*64
    float* agg  = h    + (size_t)n * F_OUT;     // n*64 (aliased as act)
    float* deg  = agg  + (size_t)n * F_OUT;     // n
    float* dinv = deg  + n;                     // n
    float* fsum = dinv + n;                     // 64
    float* fss  = fsum + 64;                    // 64

    zero_stats<<<1, 128, 0, stream>>>(fsum, fss);

    gemm_wmma_bf16<<<(n + 31) / 32, 256, 0, stream>>>(x, W, h, n);

    deg_init<<<(n + 255) / 256, 256, 0, stream>>>(deg, n);
    deg_count<<<(E + 255) / 256, 256, 0, stream>>>(ei + E, deg, E);

    {
        long total = (long)n * F_OUT;
        selfloop_init<<<(int)((total + 255) / 256), 256, 0, stream>>>(h, deg, dinv, agg, n);
    }
    {
        long total = (long)E * F_OUT;
        edge_scatter<<<(int)((total + 255) / 256), 256, 0, stream>>>(ei, h, dinv, agg, E);
    }

    finalize_stats<<<(n + 127) / 128, 256, 0, stream>>>(agg, deg, bias, fsum, fss, n);

    {
        long total = (long)n * F_OUT;
        bn_apply<<<(int)((total + 255) / 256), 256, 0, stream>>>(agg, fsum, fss, gamma, beta, y, n);
    }
}
